// MultiHeadAttention_13984413516049
// MI455X (gfx1250) — matrix operand order repaired and checked by reading, not compiled, not run on hardware
//
#include <hip/hip_runtime.h>
#include <hip/hip_bf16.h>

// MI455X (gfx1250) fused causal MHA forward.
//  k0: fp32 -> bf16 convert (x, Wq, Wk, Wv)
//  k1: QKV projection, register-blocked 32x64 per wave (8 WMMAs / 12 b128 loads
//      per k-step), bias add, V stored transposed for later B-layout loads
//  k2: flash attention, 32 queries per wave (two 16-row sub-tiles sharing K and
//      V tiles), online softmax, LDS C->A repack of P
// All matrix math: v_wmma_f32_16x16x32_bf16 (bf16 operands, fp32 accumulate).

typedef __attribute__((ext_vector_type(16))) __bf16 v16bf;
typedef __attribute__((ext_vector_type(8)))  float  v8f;

union BV { v16bf v; uint4 u[2]; };

#define S_LEN 2048
#define NH    12
#define DK    64
#define DM    768
#define NB    4
#define MROWS (NB * S_LEN)   // 8192

__device__ __forceinline__ unsigned short f2bf(float f) {
  unsigned u = __float_as_uint(f);
  unsigned r = u + 0x7FFFu + ((u >> 16) & 1u);   // round-to-nearest-even
  return (unsigned short)(r >> 16);
}

__global__ void cvt_f32_bf16(const float* __restrict__ src,
                             unsigned short* __restrict__ dst, int n) {
  int i = blockIdx.x * blockDim.x + threadIdx.x;
  int stride = gridDim.x * blockDim.x;
  for (; i < n; i += stride) dst[i] = f2bf(src[i]);
}

// ---------------- QKV projection: y = x @ W^T + b ------------------------
// One wave computes a 32x64 output block (2 m-subtiles x 4 n-subtiles).
// Per k-step of 32: 4 A b128 loads + 8 B b128 loads feed 8 WMMAs.
__global__ void __launch_bounds__(256) qkv_gemm(
    const unsigned short* __restrict__ xb,
    const unsigned short* __restrict__ wq,
    const unsigned short* __restrict__ wk,
    const unsigned short* __restrict__ wv,
    const float* __restrict__ bq, const float* __restrict__ bk,
    const float* __restrict__ bv,
    unsigned short* __restrict__ qo, unsigned short* __restrict__ ko,
    unsigned short* __restrict__ vTo)
{
  const int lane = threadIdx.x & 31;
  const int wave = threadIdx.x >> 5;
  const int nblocks = DM / 64;           // 12 (== head index)
  const int mblocks = MROWS / 32;        // 256
  int gid = blockIdx.x * 8 + wave;
  int mat = gid / (mblocks * nblocks);
  int rem = gid % (mblocks * nblocks);
  int mb = rem / nblocks;
  int nb = rem % nblocks;

  const unsigned short* W    = (mat == 0) ? wq : (mat == 1) ? wk : wv;
  const float*          bias = (mat == 0) ? bq : (mat == 1) ? bk : bv;

  const int m0 = mb * 32, n0 = nb * 64;
  const int l16 = lane & 15;
  const int hi  = lane >> 4;

  // A operand, this code's contiguous k order (B matches): lane<16 holds K 0-15; lane>=16: K 16-31
  const unsigned short* ar0 = xb + (size_t)(m0 + l16) * DM + 16 * hi;
  const unsigned short* ar1 = ar0 + (size_t)16 * DM;
  // B operand, same contiguous k order as A: lane n holds 16 contiguous K of column n (= row n of W)
  const unsigned short* br[4];
  #pragma unroll
  for (int ns = 0; ns < 4; ++ns)
    br[ns] = W + (size_t)(n0 + 16 * ns + l16) * DM + 16 * hi;

  v8f c[2][4];
  #pragma unroll
  for (int ms = 0; ms < 2; ++ms)
    #pragma unroll
    for (int ns = 0; ns < 4; ++ns) c[ms][ns] = (v8f){};

  for (int k0 = 0; k0 < DM; k0 += 32) {
    BV a0, a1;
    a0.u[0] = *(const uint4*)(ar0 + k0);
    a0.u[1] = *(const uint4*)(ar0 + k0 + 8);
    a1.u[0] = *(const uint4*)(ar1 + k0);
    a1.u[1] = *(const uint4*)(ar1 + k0 + 8);
    #pragma unroll
    for (int ns = 0; ns < 4; ++ns) {
      BV bb;
      bb.u[0] = *(const uint4*)(br[ns] + k0);
      bb.u[1] = *(const uint4*)(br[ns] + k0 + 8);
      c[0][ns] = __builtin_amdgcn_wmma_f32_16x16x32_bf16(
          false, a0.v, false, bb.v, (short)0, c[0][ns], false, false);
      c[1][ns] = __builtin_amdgcn_wmma_f32_16x16x32_bf16(
          false, a1.v, false, bb.v, (short)0, c[1][ns], false, false);
    }
  }

  const int h = nb;                       // 64-wide n-block == one head
  const int b_idx = m0 / S_LEN;
  const int m_in  = m0 % S_LEN;

  #pragma unroll
  for (int ns = 0; ns < 4; ++ns) {
    const int d = 16 * ns + l16;
    const float bval = bias[n0 + 16 * ns + l16];
    const size_t qkBase = ((size_t)(b_idx * NH + h) * S_LEN) * DK + d;
    const size_t vBase  = ((size_t)(b_idx * NH + h) * DK + d) * S_LEN;
    #pragma unroll
    for (int ms = 0; ms < 2; ++ms) {
      #pragma unroll
      for (int i = 0; i < 8; ++i) {
        int row = m_in + 16 * ms + i + 8 * hi;   // C layout: lanes>=16 -> rows+8
        unsigned short e = f2bf(c[ms][ns][i] + bval);
        if (mat == 0)      qo[qkBase + (size_t)row * DK] = e;
        else if (mat == 1) ko[qkBase + (size_t)row * DK] = e;
        else               vTo[vBase + row] = e;  // V transposed [d_k, S]
      }
    }
  }
}

// ---------------- Flash attention, causal -------------------------------
// One wave owns 32 queries (two 16-row sub-tiles); loops key blocks of 32.
// K tiles and V^T tiles are loaded once per block and shared by both sub-tiles.
__global__ void __launch_bounds__(256) flash_attn(
    const unsigned short* __restrict__ qb,
    const unsigned short* __restrict__ kb,
    const unsigned short* __restrict__ vTb,
    float* __restrict__ out)
{
  __shared__ unsigned short ldsP[8][2][16 * 32];  // per-wave/sub-tile P staging
  const int lane = threadIdx.x & 31;
  const int wave = threadIdx.x >> 5;
  const int bh = blockIdx.y;
  const int b = bh / NH, h = bh % NH;
  const int q0 = blockIdx.x * 256 + wave * 32;
  const int l16 = lane & 15;
  const int hi  = lane >> 4;

  const unsigned short* Q  = qb  + (size_t)bh * S_LEN * DK;
  const unsigned short* K  = kb  + (size_t)bh * S_LEN * DK;
  const unsigned short* VT = vTb + (size_t)bh * DK * S_LEN;

  // Q sub-tiles in A layout, two d-halves each
  BV qa[2][2];
  #pragma unroll
  for (int qs = 0; qs < 2; ++qs) {
    const unsigned short* qrow = Q + (size_t)(q0 + 16 * qs + l16) * DK + 16 * hi;
    #pragma unroll
    for (int dh = 0; dh < 2; ++dh) {
      qa[qs][dh].u[0] = *(const uint4*)(qrow + 32 * dh);
      qa[qs][dh].u[1] = *(const uint4*)(qrow + 32 * dh + 8);
    }
  }

  v8f o[2][4];
  float mrow[2][8], lrow[2][8];
  #pragma unroll
  for (int qs = 0; qs < 2; ++qs) {
    #pragma unroll
    for (int t = 0; t < 4; ++t) o[qs][t] = (v8f){};
    #pragma unroll
    for (int i = 0; i < 8; ++i) { mrow[qs][i] = -3.0e38f; lrow[qs][i] = 0.0f; }
  }

  for (int kblk = 0; kblk <= q0 + 31; kblk += 32) {
    const bool doq0 = kblk <= q0 + 15;   // lower sub-tile past its diagonal?

    // ---- K tiles: 2 key sub-tiles x 2 d-halves, shared by both q sub-tiles
    BV kt[2][2];
    #pragma unroll
    for (int ts = 0; ts < 2; ++ts) {
      #pragma unroll
      for (int dh = 0; dh < 2; ++dh) {
        const unsigned short* kr =
            K + (size_t)(kblk + 16 * ts + l16) * DK + 32 * dh + 16 * hi;
        kt[ts][dh].u[0] = *(const uint4*)(kr);
        kt[ts][dh].u[1] = *(const uint4*)(kr + 8);
      }
    }

    // ---- per query sub-tile: scores, online softmax, stage P in LDS
    #pragma unroll
    for (int qs = 0; qs < 2; ++qs) {
      if (qs == 0 && !doq0) continue;    // wave-uniform skip

      v8f s0 = (v8f){}, s1 = (v8f){};
      #pragma unroll
      for (int dh = 0; dh < 2; ++dh) {
        s0 = __builtin_amdgcn_wmma_f32_16x16x32_bf16(
            false, qa[qs][dh].v, false, kt[0][dh].v, (short)0, s0, false, false);
        s1 = __builtin_amdgcn_wmma_f32_16x16x32_bf16(
            false, qa[qs][dh].v, false, kt[1][dh].v, (short)0, s1, false, false);
      }

      const int qrb = q0 + 16 * qs + 8 * hi;
      const bool need_mask = (kblk + 31) > (q0 + 16 * qs);
      #pragma unroll
      for (int i = 0; i < 8; ++i) {
        s0[i] *= 0.125f;                 // 1/sqrt(64)
        s1[i] *= 0.125f;
        if (need_mask) {
          int qi = qrb + i;
          if (kblk + l16 > qi)      s0[i] = -1.0e9f;
          if (kblk + 16 + l16 > qi) s1[i] = -1.0e9f;
        }
      }

      unsigned short* pl = ldsP[wave][qs];
      #pragma unroll
      for (int i = 0; i < 8; ++i) {
        float v = fmaxf(s0[i], s1[i]);
        v = fmaxf(v, __shfl_xor(v, 1, 32));
        v = fmaxf(v, __shfl_xor(v, 2, 32));
        v = fmaxf(v, __shfl_xor(v, 4, 32));
        v = fmaxf(v, __shfl_xor(v, 8, 32));
        float mn = fmaxf(mrow[qs][i], v);
        float alpha = __expf(mrow[qs][i] - mn);
        float p0 = __expf(s0[i] - mn);
        float p1 = __expf(s1[i] - mn);
        float r = p0 + p1;
        r += __shfl_xor(r, 1, 32);
        r += __shfl_xor(r, 2, 32);
        r += __shfl_xor(r, 4, 32);
        r += __shfl_xor(r, 8, 32);
        lrow[qs][i] = lrow[qs][i] * alpha + r;
        mrow[qs][i] = mn;
        o[qs][0][i] *= alpha; o[qs][1][i] *= alpha;
        o[qs][2][i] *= alpha; o[qs][3][i] *= alpha;
        pl[(i + 8 * hi) * 32 + l16]      = f2bf(p0);   // C layout -> row-major
        pl[(i + 8 * hi) * 32 + l16 + 16] = f2bf(p1);
      }
    }
    asm volatile("s_wait_dscnt 0x0" ::: "memory");     // wave-local LDS handoff

    // ---- reload P in WMMA A layout for both sub-tiles
    BV pa[2];
    #pragma unroll
    for (int qs = 0; qs < 2; ++qs) {
      if (qs == 0 && !doq0) continue;
      const unsigned short* pr = ldsP[wave][qs] + l16 * 32 + 16 * hi;
      pa[qs].u[0] = *(const uint4*)(pr);
      pa[qs].u[1] = *(const uint4*)(pr + 8);
    }

    // ---- O += P * V : each V^T tile loaded once, feeds both sub-tiles
    #pragma unroll
    for (int t = 0; t < 4; ++t) {
      BV vv;
      const unsigned short* vr =
          VT + (size_t)(16 * t + l16) * S_LEN + kblk + 16 * hi;
      vv.u[0] = *(const uint4*)(vr);
      vv.u[1] = *(const uint4*)(vr + 8);
      if (doq0)
        o[0][t] = __builtin_amdgcn_wmma_f32_16x16x32_bf16(
            false, pa[0].v, false, vv.v, (short)0, o[0][t], false, false);
      o[1][t] = __builtin_amdgcn_wmma_f32_16x16x32_bf16(
          false, pa[1].v, false, vv.v, (short)0, o[1][t], false, false);
    }
  }

  // ---- epilogue: normalize and store fp32 [B, S, d_model]
  #pragma unroll
  for (int qs = 0; qs < 2; ++qs) {
    #pragma unroll
    for (int i = 0; i < 8; ++i) {
      float inv = 1.0f / lrow[qs][i];
      int q = q0 + 16 * qs + i + 8 * hi;
      float* orow = out + ((size_t)(b * S_LEN + q)) * DM + h * DK + l16;
      orow[0]  = o[qs][0][i] * inv;
      orow[16] = o[qs][1][i] * inv;
      orow[32] = o[qs][2][i] * inv;
      orow[48] = o[qs][3][i] * inv;
    }
  }
}

extern "C" void kernel_launch(void* const* d_in, const int* in_sizes, int n_in,
                              void* d_out, int out_size, void* d_ws, size_t ws_size,
                              hipStream_t stream) {
  const float* x  = (const float*)d_in[0];
  // d_in[1] = causal mask (computed analytically, unused)
  const float* Wq = (const float*)d_in[2];
  const float* bq = (const float*)d_in[3];
  const float* Wk = (const float*)d_in[4];
  const float* bk = (const float*)d_in[5];
  const float* Wv = (const float*)d_in[6];
  const float* bv = (const float*)d_in[7];
  float* out = (float*)d_out;

  char* ws = (char*)d_ws;
  unsigned short* xb  = (unsigned short*)(ws);                 // 12.58 MB
  unsigned short* wqb = (unsigned short*)(ws + 12582912);      // 1.18 MB
  unsigned short* wkb = (unsigned short*)(ws + 13762560);
  unsigned short* wvb = (unsigned short*)(ws + 14942208);
  unsigned short* qbf = (unsigned short*)(ws + 16121856);      // 12.58 MB
  unsigned short* kbf = (unsigned short*)(ws + 28704768);
  unsigned short* vTb = (unsigned short*)(ws + 41287680);      // total ~54 MB

  const int nx = MROWS * DM;     // 6.29M
  const int nw = DM * DM;        // 0.59M
  cvt_f32_bf16<<<512, 256, 0, stream>>>(x,  xb,  nx);
  cvt_f32_bf16<<<256, 256, 0, stream>>>(Wq, wqb, nw);
  cvt_f32_bf16<<<256, 256, 0, stream>>>(Wk, wkb, nw);
  cvt_f32_bf16<<<256, 256, 0, stream>>>(Wv, wvb, nw);

  // 3 mats * 256 m-blocks * 12 n-blocks = 9216 wave-blocks / 8 waves per block
  qkv_gemm<<<1152, 256, 0, stream>>>(xb, wqb, wkb, wvb, bq, bk, bv,
                                     qbf, kbf, vTb);

  // grid: 8 query-blocks of 256 x (B*H = 48)
  flash_attn<<<dim3(8, 48), 256, 0, stream>>>(qbf, kbf, vTb, out);
}
